// EFAdLIF_8933531976415
// MI455X (gfx1250) — matrix operand order repaired and checked by reading, not compiled, not run on hardware
//
#include <hip/hip_runtime.h>
#include <hip/hip_bf16.h>

typedef __attribute__((ext_vector_type(16))) __bf16 v16bf;
typedef __attribute__((ext_vector_type(8)))  __bf16 v8bf;
typedef __attribute__((ext_vector_type(8)))  float  v8f;
typedef __attribute__((ext_vector_type(4)))  unsigned int u32x4;
typedef __attribute__((ext_vector_type(8)))  unsigned int u32x8;

#define TSTEPS 2000
#define BATCH  32
#define NIN    256
#define NHID   512
#define MROWS  (BATCH * TSTEPS)      // 64000 rows of the input GEMM
#define QSCALE 120.0f
#define SCAN_BLOCKS 64               // 2 batch-tiles x 32 n-tiles

// ---------------------------------------------------------------------------
// Fragment loaders matching CDNA5 WMMA VGPR layouts (CDNA5 ISA 7.12.2)
// ---------------------------------------------------------------------------

// A-matrix 16x32 bf16 from row-major [16+][ldk]:
// lane L: M = L&15, khalf = (L>>4)*8; elems 0..7 -> K=k0+khalf+e,
// elems 8..15 -> K=k0+khalf+16+(e-8).  Two contiguous 16B loads.
__device__ __forceinline__ v16bf load_frag_a(const __bf16* base, int ldk,
                                             int m0, int k0, int lane) {
    const int m  = m0 + (lane & 15);
    const int kh = (lane >> 4) << 3;
    const __bf16* p = base + (long)m * ldk + k0 + kh;
    v8bf lo = *(const v8bf*)(p);
    v8bf hi = *(const v8bf*)(p + 16);
    v16bf a;
#pragma unroll
    for (int e = 0; e < 8; ++e) { a[e] = lo[e]; a[e + 8] = hi[e]; }
    return a;
}

// B-matrix 32x16 bf16, B[k][n] taken from row-major mat[n][k] (no transpose
// needed: W[n][i] and R[n][k] both give contiguous k).
// lane L: N = n0+(L&15), Kbase = k0+(L>>4)*8; elems 0..7 -> K = Kbase+e,
// elems 8..15 -> K = Kbase+16+(e-8).
__device__ __forceinline__ v16bf load_frag_b(const __bf16* mat, int ldk,
                                             int n0, int k0, int lane) {
    const int n  = n0 + (lane & 15);
    const int kb = k0 + ((lane >> 4) << 3);
    const __bf16* p = mat + (long)n * ldk + kb;
    v8bf lo = *(const v8bf*)(p);
    v8bf hi = *(const v8bf*)(p + 16);
    v16bf b;
#pragma unroll
    for (int e = 0; e < 8; ++e) { b[e] = lo[e]; b[e + 8] = hi[e]; }
    return b;
}

// ---------------------------------------------------------------------------
// fp32 -> bf16 conversion of x, weight, recurrent (grid-stride)
// ---------------------------------------------------------------------------
__global__ void k_convert(const float* __restrict__ x,
                          const float* __restrict__ w,
                          const float* __restrict__ r,
                          __bf16* __restrict__ xb,
                          __bf16* __restrict__ wb,
                          __bf16* __restrict__ rb) {
    const long i = (long)blockIdx.x * blockDim.x + threadIdx.x;
    const long stride = (long)gridDim.x * blockDim.x;
    const long nx = (long)MROWS * NIN;
    for (long j = i; j < nx; j += stride)           xb[j] = (__bf16)x[j];
    for (long j = i; j < NHID * NIN; j += stride)   wb[j] = (__bf16)w[j];
    for (long j = i; j < NHID * NHID; j += stride)  rb[j] = (__bf16)r[j];
}

// Zero the z ping-pong buffers and the per-step barrier counters each launch
// (workspace is poisoned once and never restored between replays).
__global__ void k_zero(__bf16* __restrict__ zb, unsigned* __restrict__ cnt) {
    const int i = blockIdx.x * blockDim.x + threadIdx.x;
    if (i < 2 * BATCH * NHID) zb[i] = (__bf16)0.0f;
    if (i < TSTEPS) cnt[i] = 0u;
}

// ---------------------------------------------------------------------------
// Phase 1: cur[m][n] = sum_i x[m][i]*W[n][i] + bias[n]
// block = 4 waves, covers one 16-row M tile x four 16-col N tiles.
// A tile (16x256 bf16 = 8KB) DMA'd into LDS by the Tensor Data Mover
// (tensor_load_to_lds, 2D descriptor), shared by the 4 waves.
// ---------------------------------------------------------------------------
__global__ void __launch_bounds__(128) k_gemm_in(const __bf16* __restrict__ xb,
                                                 const __bf16* __restrict__ wb,
                                                 const float* __restrict__ bias,
                                                 float* __restrict__ cur) {
    extern __shared__ __align__(16) __bf16 atile[];   // 16*NIN bf16, LDS offset 0

    const int tid  = threadIdx.x;
    const int lane = tid & 31;
    const int wave = tid >> 5;
    const int m0   = blockIdx.x * 16;                 // 0..63984
    const int n0   = (blockIdx.y * 4 + wave) * 16;    // 0..496

    // TDM: DMA 16 rows x 256 bf16 (8KB) of x into LDS. D# per ISA 08 §8.3/8.4.
    if (wave == 0) {
        const unsigned long long ga = (unsigned long long)(xb + (long)m0 * NIN);
        u32x4 g0;
        g0[0] = 1u;                                          // count=1 user D#
        g0[1] = 0u;                                          // lds_addr = 0
        g0[2] = (unsigned)(ga & 0xFFFFFFFFu);                // global_addr lo
        g0[3] = (unsigned)((ga >> 32) & 0x01FFFFFFu)         // global_addr hi
              | (2u << 30);                                  // type = 2 (image)
        u32x8 g1;
        g1[0] = 1u << 16;                                    // data_size=1 (2B)
        g1[1] = ((unsigned)NIN & 0xFFFFu) << 16;             // tensor_dim0 lo16
        g1[2] = ((unsigned)NIN >> 16)                        // tensor_dim0 hi16
              | (((unsigned)MROWS & 0xFFFFu) << 16);         // tensor_dim1 lo16
        g1[3] = ((unsigned)MROWS >> 16)                      // tensor_dim1 hi16
              | ((unsigned)NIN << 16);                       // tile_dim0 = 256
        g1[4] = 16u;                                         // tile_dim1=16, tile_dim2=0
        g1[5] = (unsigned)NIN;                               // tensor_dim0_stride lo32
        g1[6] = 0u;                                          // stride hi / dim1_stride lo
        g1[7] = 0u;
        asm volatile("tensor_load_to_lds %0, %1" :: "s"(g0), "s"(g1) : "memory");
        __builtin_amdgcn_s_wait_tensorcnt(0);
    }
    __syncthreads();

    v8f c = {};
#pragma unroll
    for (int k0 = 0; k0 < NIN; k0 += 32) {
        if (k0 + 32 < NIN)
            __builtin_prefetch(wb + (long)(n0 + (lane & 15)) * NIN + k0 + 32, 0, 0);
        v16bf a = load_frag_a(atile, NIN, 0, k0, lane);
        v16bf b = load_frag_b(wb, NIN, n0, k0, lane);
        c = __builtin_amdgcn_wmma_f32_16x16x32_bf16(false, a, false, b,
                                                    (short)0, c, false, false);
    }

    const int n = n0 + (lane & 15);
    const float bv = bias[n];
#pragma unroll
    for (int r = 0; r < 8; ++r) {
        const int m = m0 + r + ((lane >> 4) << 3);
        cur[(long)m * NHID + n] = c[r] + bv;
    }
}

// ---------------------------------------------------------------------------
// Phase 2: persistent AdLIF scan. 64 blocks x 1 wave; one 16x16 (batch x n)
// tile per wave. Recurrent-weight B fragments are time-invariant -> hoisted
// into 128 VGPRs before the t-loop. z state ping-pongs through L2 in bf16;
// u/w/z_prev live in registers. Grid-wide barrier per step.
// ---------------------------------------------------------------------------
__global__ void __launch_bounds__(32) k_scan(const float* __restrict__ cur,
                                             const __bf16* __restrict__ rb,
                                             const float* __restrict__ du,
                                             const float* __restrict__ dw,
                                             const float* __restrict__ aa,
                                             const float* __restrict__ bb,
                                             __bf16* __restrict__ zbuf,
                                             unsigned* __restrict__ cnt,
                                             float* __restrict__ out) {
    const int lane = threadIdx.x;
    const int tile = blockIdx.x;          // 0..63
    const int m0   = (tile & 1) * 16;     // batch tile
    const int n0   = (tile >> 1) * 16;    // hidden tile
    const int n    = n0 + (lane & 15);

    const float duv = du[n], dwv = dw[n], av = aa[n], bv = bb[n];

    // Hoist recurrent-weight fragments: R[n][k], 16 K-tiles of 32 -> 16 frags.
    v16bf bfr[NHID / 32];
#pragma unroll
    for (int kt = 0; kt < NHID / 32; ++kt)
        bfr[kt] = load_frag_b(rb, NHID, n0, kt * 32, lane);

    float u[8], w[8], zp[8];
#pragma unroll
    for (int r = 0; r < 8; ++r) { u[r] = 0.f; w[r] = 0.f; zp[r] = 0.f; }

    for (int t = 0; t < TSTEPS; ++t) {
        const __bf16* zprev = zbuf + (t & 1) * (BATCH * NHID);
        __bf16*       znext = zbuf + ((t + 1) & 1) * (BATCH * NHID);

        // acc = z_{t-1} @ R^T   (A from L2-resident z buffer, B from VGPRs)
        v8f acc = {};
#pragma unroll
        for (int kt = 0; kt < NHID / 32; ++kt) {
            v16bf a = load_frag_a(zprev, NHID, m0, kt * 32, lane);
            acc = __builtin_amdgcn_wmma_f32_16x16x32_bf16(false, a, false, bfr[kt],
                                                          (short)0, acc, false, false);
        }

#pragma unroll
        for (int r = 0; r < 8; ++r) {
            const int  b_  = m0 + r + ((lane >> 4) << 3);
            const long row = (long)b_ * TSTEPS + t;
            const float ct = cur[row * NHID + n] + acc[r];
            const float un = duv * u[r] + (1.f - duv) * (ct - w[r]);
            const float z  = (un - 1.0f) > 0.f ? 1.f : 0.f;
            const float wn = dwv * w[r] + (1.f - dwv) * (av * u[r] + bv * zp[r]) * QSCALE;
            u[r]  = un * (1.f - z);       // reset to u_rest = 0
            w[r]  = wn;
            zp[r] = z;
            out[row * NHID + n]       = z;
            znext[b_ * NHID + n]      = (__bf16)z;
        }

        // Cluster barrier (HW NOP when not cluster-dispatched) + grid barrier.
        __builtin_amdgcn_s_cluster_barrier();
        __threadfence();
        if (lane == 0) {
            __hip_atomic_fetch_add(&cnt[t], 1u, __ATOMIC_RELEASE,
                                   __HIP_MEMORY_SCOPE_AGENT);
            while (__hip_atomic_load(&cnt[t], __ATOMIC_ACQUIRE,
                                     __HIP_MEMORY_SCOPE_AGENT) < (unsigned)SCAN_BLOCKS)
                __builtin_amdgcn_s_sleep(2);
        }
        __builtin_amdgcn_wave_barrier();  // whole wave waits with lane 0
        __threadfence();
    }
}

// ---------------------------------------------------------------------------
extern "C" void kernel_launch(void* const* d_in, const int* in_sizes, int n_in,
                              void* d_out, int out_size, void* d_ws, size_t ws_size,
                              hipStream_t stream) {
    const float* x      = (const float*)d_in[0];
    const float* weight = (const float*)d_in[1];
    const float* bias   = (const float*)d_in[2];
    const float* rec    = (const float*)d_in[3];
    const float* du     = (const float*)d_in[4];
    const float* dw     = (const float*)d_in[5];
    const float* aa     = (const float*)d_in[6];
    const float* bb     = (const float*)d_in[7];
    float* out = (float*)d_out;

    char* ws = (char*)d_ws;
    size_t off = 0;
    auto take = [&](size_t bytes) -> void* {
        void* p = ws + off;
        off = (off + bytes + 255) & ~(size_t)255;
        return p;
    };
    __bf16*   xb   = (__bf16*)take((size_t)MROWS * NIN * sizeof(__bf16));
    __bf16*   wb   = (__bf16*)take((size_t)NHID * NIN * sizeof(__bf16));
    __bf16*   rb   = (__bf16*)take((size_t)NHID * NHID * sizeof(__bf16));
    float*    cur  = (float*)take((size_t)MROWS * NHID * sizeof(float));
    __bf16*   zbuf = (__bf16*)take((size_t)2 * BATCH * NHID * sizeof(__bf16));
    unsigned* cnt  = (unsigned*)take((size_t)TSTEPS * sizeof(unsigned));
    (void)ws_size; (void)in_sizes; (void)n_in; (void)out_size;

    // 1) convert fp32 -> bf16
    k_convert<<<2048, 256, 0, stream>>>(x, weight, rec, xb, wb, rb);

    // 2) zero z state + barrier counters (ws is poisoned, not restored)
    const int nz = 2 * BATCH * NHID;           // 32768 >= TSTEPS
    k_zero<<<(nz + 255) / 256, 256, 0, stream>>>(zbuf, cnt);

    // 3) input projection GEMM: 4000 M-tiles x 8 N-groups, 4 waves/block,
    //    8KB dynamic LDS for the TDM-staged A tile
    k_gemm_in<<<dim3(MROWS / 16, NHID / 64), 128,
                16 * NIN * sizeof(__bf16), stream>>>(xb, wb, bias, cur);

    // 4) persistent recurrent scan: 64 single-wave blocks
    k_scan<<<SCAN_BLOCKS, 32, 0, stream>>>(cur, rb, du, dw, aa, bb, zbuf, cnt, out);
}
